// AttnDecoderRNN_29008209117241
// MI455X (gfx1250) — matrix operand order repaired and checked by reading, not compiled, not run on hardware
//
#include <hip/hip_runtime.h>
#include <math.h>

// ---- problem constants ----
#define Bz   32
#define Tz   128
#define Hz   512
#define Vz   32000
#define Sz   64      // BLOCK decode steps

typedef __attribute__((ext_vector_type(16))) __bf16 v16bf;
typedef __attribute__((ext_vector_type(8)))  __bf16 v8bf;
typedef __attribute__((ext_vector_type(8)))  float  v8f;

// ---------------- WMMA helpers (CDNA5 bf16 16x16x32) ----------------
__device__ __forceinline__ v8f wmma_bf16(v16bf a, v16bf b, v8f c) {
    // 8 args: (neg_a, A, neg_b, B, c_mod, C, reuse_a, reuse_b)
    return __builtin_amdgcn_wmma_f32_16x16x32_bf16(false, a, false, b, (short)0, c, false, false);
}

// A matrix 16x32 bf16, row-major src[m][k], ld = row stride in elements.
// lane L: m = L%16, half = L/16; elems 0..7 -> k = kk + half*8 + e ; elems 8..15 -> +16
__device__ __forceinline__ v16bf load_a_frag(const __bf16* src, int ld, int mrow, int kk, int lane) {
    int lhalf = lane >> 4;
    const __bf16* p = src + (size_t)mrow * ld + kk + lhalf * 8;
    v8bf lo = *(const v8bf*)(p);
    v8bf hi = *(const v8bf*)(p + 16);
    v16bf a;
#pragma unroll
    for (int i = 0; i < 8; ++i) { a[i] = lo[i]; a[i + 8] = hi[i]; }
    return a;
}

// B matrix 32x16 bf16 where B[k][n] = W[nrow][k], W row-major [N][K], ld = K stride.
// lane L: n = L%16, half = L/16; elems 0..7 -> k = kk + half*8 + e ; elems 8..15 -> +16
__device__ __forceinline__ v16bf load_b_frag(const __bf16* W, int ld, int nrow, int kk, int lane) {
    int lhalf = lane >> 4;
    const __bf16* p = W + (size_t)nrow * ld + kk + lhalf * 8;
    v8bf lo = *(const v8bf*)(p);
    v8bf hi = *(const v8bf*)(p + 16);
    v16bf b;
#pragma unroll
    for (int i = 0; i < 8; ++i) { b[i] = lo[i]; b[i + 8] = hi[i]; }
    return b;
}

__device__ __forceinline__ float sigf(float x) { return 1.0f / (1.0f + __expf(-x)); }

// ---------------- utility kernels ----------------
__global__ void k_cvt_bf16(const float* __restrict__ src, __bf16* __restrict__ dst, int n) {
    for (int i = blockIdx.x * blockDim.x + threadIdx.x; i < n; i += gridDim.x * blockDim.x)
        dst[i] = (__bf16)src[i];
}

__global__ void k_init_h(const float* __restrict__ eh, float* __restrict__ h0, __bf16* __restrict__ h0bf) {
    int i = blockIdx.x * blockDim.x + threadIdx.x;   // 16384 elems
    float v = eh[i];
    h0[i] = v;
    h0bf[i] = (__bf16)v;
}

__global__ void k_copy_f32(const float* __restrict__ src, float* __restrict__ dst, int n) {
    for (int i = blockIdx.x * blockDim.x + threadIdx.x; i < n; i += gridDim.x * blockDim.x)
        dst[i] = src[i];
}

// ---------------- phase 1: ua_keys[b,t,o] = enc[b,t,:] . Ua_w[o,:] + Ua_b[o] ----------------
__global__ void k_uakeys(const float* __restrict__ enc, const float* __restrict__ Ua_w,
                         const float* __restrict__ Ua_b, float* __restrict__ ua_keys) {
    __shared__ float sh[Hz];
    int row = blockIdx.x;            // b*T + t  (0..4095)
    int tid = threadIdx.x;           // 256 threads
    sh[tid]       = enc[(size_t)row * Hz + tid];
    sh[tid + 256] = enc[(size_t)row * Hz + tid + 256];
    __syncthreads();
#pragma unroll
    for (int rep = 0; rep < 2; ++rep) {
        int o = tid + rep * 256;
        const float* w = Ua_w + (size_t)o * Hz;
        float acc = Ua_b[o];
#pragma unroll 8
        for (int k = 0; k < Hz; ++k) acc = fmaf(w[k], sh[k], acc);
        ua_keys[(size_t)row * Hz + o] = acc;
    }
}

// ---------------- phase 2a: per-batch attention + build x = [emb, ctx] ----------------
__global__ void k_attn(const float* __restrict__ enc, const float* __restrict__ ua_keys,
                       const float* __restrict__ h_cur,
                       const float* __restrict__ Wa_w, const float* __restrict__ Wa_b,
                       const float* __restrict__ Va_w, const float* __restrict__ Va_b,
                       const float* __restrict__ emb, const int* __restrict__ target,
                       __bf16* __restrict__ x_bf, float* __restrict__ attn_out, int s) {
    __shared__ float sh_h[Hz];
    __shared__ float sh_q[Hz];
    __shared__ float sh_r[256];
    __shared__ float sh_w[Tz];
    int b = blockIdx.x, tid = threadIdx.x;

    sh_h[tid]       = h_cur[b * Hz + tid];
    sh_h[tid + 256] = h_cur[b * Hz + tid + 256];
    __syncthreads();

    // q = Wa_w . h + Wa_b
#pragma unroll
    for (int rep = 0; rep < 2; ++rep) {
        int o = tid + rep * 256;
        const float* w = Wa_w + (size_t)o * Hz;
        float acc = Wa_b[o];
#pragma unroll 8
        for (int k = 0; k < Hz; ++k) acc = fmaf(w[k], sh_h[k], acc);
        sh_q[o] = acc;
    }
    __syncthreads();

    // scores[t] = sum_k tanh(q[k] + ua_keys[b,t,k]) * Va_w[k]   (2 threads per t)
    {
        int t = tid >> 1, half = tid & 1;
        const float* uk = ua_keys + ((size_t)b * Tz + t) * Hz + half * 256;
        const float* q  = sh_q + half * 256;
        const float* va = Va_w + half * 256;
        float acc = 0.f;
#pragma unroll 4
        for (int k = 0; k < 256; ++k) acc += tanhf(q[k] + uk[k]) * va[k];
        sh_r[tid] = acc;
    }
    __syncthreads();
    if (tid < Tz) sh_w[tid] = sh_r[2 * tid] + sh_r[2 * tid + 1] + Va_b[0];
    __syncthreads();

    // softmax over T=128
    float v = (tid < Tz) ? sh_w[tid] : -INFINITY;
    sh_r[tid] = v;
    __syncthreads();
    for (int st = 128; st >= 1; st >>= 1) {
        if (tid < st) sh_r[tid] = fmaxf(sh_r[tid], sh_r[tid + st]);
        __syncthreads();
    }
    float mx = sh_r[0];
    __syncthreads();
    float e = (tid < Tz) ? __expf(sh_w[tid] - mx) : 0.f;
    sh_r[tid] = e;
    __syncthreads();
    for (int st = 128; st >= 1; st >>= 1) {
        if (tid < st) sh_r[tid] += sh_r[tid + st];
        __syncthreads();
    }
    float inv = 1.0f / sh_r[0];
    __syncthreads();
    if (tid < Tz) {
        float w = e * inv;
        sh_w[tid] = w;
        attn_out[((size_t)b * Sz + s) * Tz + tid] = w;
    }
    __syncthreads();

    // ctx[o] = sum_t w[t] * enc[b,t,o];  x = [emb(token), ctx] in bf16
    int token = (s == 0) ? 1 : target[b * Sz + (s - 1)];
    const float* erow = emb + (size_t)token * Hz;
#pragma unroll
    for (int rep = 0; rep < 2; ++rep) {
        int o = tid + rep * 256;
        float acc = 0.f;
#pragma unroll 4
        for (int t = 0; t < Tz; ++t) acc = fmaf(sh_w[t], enc[((size_t)b * Tz + t) * Hz + o], acc);
        x_bf[(size_t)b * (2 * Hz) + o]      = (__bf16)erow[o];
        x_bf[(size_t)b * (2 * Hz) + Hz + o] = (__bf16)acc;
    }
}

// ---------------- phase 2b: GRU cell via WMMA bf16 (M=32, N-slice=32, gates fused) ----------------
__global__ void k_gru(const __bf16* __restrict__ x_bf, const __bf16* __restrict__ h_bf,
                      const float* __restrict__ h_cur,
                      const __bf16* __restrict__ Wih, const __bf16* __restrict__ Whh,
                      const float* __restrict__ b_ih, const float* __restrict__ b_hh,
                      float* __restrict__ h_next, __bf16* __restrict__ hbf_next,
                      __bf16* __restrict__ h_seq, int s) {
    int tid = threadIdx.x;               // 128 threads = 4 waves
    int wave = tid >> 5, lane = tid & 31;
    int m_tile = wave & 1;               // M = 32 -> 2 tiles of 16
    int nt = wave >> 1;                  // 2 n-tiles of 16 per block
    int n_base = blockIdx.x * 32 + nt * 16;
    int lm = lane & 15, lhalf = lane >> 4;
    int mrow = m_tile * 16 + lm;

    v8f air = {}, aiz = {}, ain = {}, ahr = {}, ahz = {}, ahn = {};

    // gi = x @ W_ih^T   (K = 1024)
    for (int kk = 0; kk < 2 * Hz; kk += 32) {
        v16bf a  = load_a_frag(x_bf, 2 * Hz, mrow, kk, lane);
        v16bf br = load_b_frag(Wih, 2 * Hz, 0 * Hz + n_base + lm, kk, lane);
        v16bf bz = load_b_frag(Wih, 2 * Hz, 1 * Hz + n_base + lm, kk, lane);
        v16bf bn = load_b_frag(Wih, 2 * Hz, 2 * Hz + n_base + lm, kk, lane);
        air = wmma_bf16(a, br, air);
        aiz = wmma_bf16(a, bz, aiz);
        ain = wmma_bf16(a, bn, ain);
    }
    // gh = h @ W_hh^T   (K = 512)
    for (int kk = 0; kk < Hz; kk += 32) {
        v16bf a  = load_a_frag(h_bf, Hz, mrow, kk, lane);
        v16bf br = load_b_frag(Whh, Hz, 0 * Hz + n_base + lm, kk, lane);
        v16bf bz = load_b_frag(Whh, Hz, 1 * Hz + n_base + lm, kk, lane);
        v16bf bn = load_b_frag(Whh, Hz, 2 * Hz + n_base + lm, kk, lane);
        ahr = wmma_bf16(a, br, ahr);
        ahz = wmma_bf16(a, bz, ahz);
        ahn = wmma_bf16(a, bn, ahn);
    }

    // epilogue: r,z,n gates + hidden update.  D layout: m = vgpr + 8*(lane>=16), n = lane%16
    int o = n_base + lm;
    float bir = b_ih[o], biz = b_ih[Hz + o], bin = b_ih[2 * Hz + o];
    float bhr = b_hh[o], bhz = b_hh[Hz + o], bhn = b_hh[2 * Hz + o];
#pragma unroll
    for (int i = 0; i < 8; ++i) {
        int m = m_tile * 16 + 8 * lhalf + i;
        float r  = sigf(air[i] + bir + ahr[i] + bhr);
        float z  = sigf(aiz[i] + biz + ahz[i] + bhz);
        float nn = tanhf(ain[i] + bin + r * (ahn[i] + bhn));
        float ho = h_cur[m * Hz + o];
        float hn = (1.0f - z) * nn + z * ho;
        h_next[m * Hz + o]   = hn;
        hbf_next[m * Hz + o] = (__bf16)hn;
        h_seq[((size_t)s * Bz + m) * Hz + o] = (__bf16)hn;
    }
}

// ---------------- phase 3: logits GEMM [2048 x 32000 x 512] via WMMA bf16 ----------------
__global__ void k_logits(const __bf16* __restrict__ h_seq, const __bf16* __restrict__ outw,
                         const float* __restrict__ out_b, float* __restrict__ dec) {
    int tid = threadIdx.x;               // 256 threads = 8 waves
    int wave = tid >> 5, lane = tid & 31;
    int m_tile = wave & 1;               // block tile M32 x N64
    int nt = wave >> 1;                  // 4 n-tiles
    int n_base = blockIdx.x * 64 + nt * 16;
    int m_base = blockIdx.y * 32 + m_tile * 16;
    int lm = lane & 15, lhalf = lane >> 4;
    int mrow = m_base + lm;
    int nrow = n_base + lm;

    v8f acc = {};
#pragma unroll 4
    for (int kk = 0; kk < Hz; kk += 32) {
        __builtin_prefetch(outw + (size_t)nrow * Hz + kk + 64, 0, 1);
        v16bf a = load_a_frag(h_seq, Hz, mrow, kk, lane);
        v16bf b = load_b_frag(outw, Hz, nrow, kk, lane);
        acc = wmma_bf16(a, b, acc);
    }

    int n = n_base + lm;
    float bias = out_b[n];
#pragma unroll
    for (int i = 0; i < 8; ++i) {
        int m = m_base + 8 * lhalf + i;  // m = s*32 + b
        int bb = m & (Bz - 1);
        int ss = m >> 5;
        dec[((size_t)bb * Sz + ss) * Vz + n] = acc[i] + bias;
    }
}

// ---------------- phase 4: in-place row-wise log_softmax over V=32000 ----------------
__global__ void k_logsoftmax(float* __restrict__ dec) {
    __shared__ float red[256];
    size_t base = (size_t)blockIdx.x * Vz;   // 2048 rows
    int tid = threadIdx.x;

    float mx = -INFINITY;
    for (int i = tid; i < Vz; i += 256) mx = fmaxf(mx, dec[base + i]);
    red[tid] = mx;
    __syncthreads();
    for (int st = 128; st >= 1; st >>= 1) {
        if (tid < st) red[tid] = fmaxf(red[tid], red[tid + st]);
        __syncthreads();
    }
    mx = red[0];
    __syncthreads();

    float sum = 0.f;
    for (int i = tid; i < Vz; i += 256) sum += __expf(dec[base + i] - mx);
    red[tid] = sum;
    __syncthreads();
    for (int st = 128; st >= 1; st >>= 1) {
        if (tid < st) red[tid] += red[tid + st];
        __syncthreads();
    }
    float lz = mx + __logf(red[0]);
    __syncthreads();

    for (int i = tid; i < Vz; i += 256) dec[base + i] -= lz;
}

// ---------------- host launcher ----------------
extern "C" void kernel_launch(void* const* d_in, const int* in_sizes, int n_in,
                              void* d_out, int out_size, void* d_ws, size_t ws_size,
                              hipStream_t stream) {
    const float* enc   = (const float*)d_in[0];
    const float* eh    = (const float*)d_in[1];
    const int*   tgt   = (const int*)  d_in[2];
    const float* emb   = (const float*)d_in[3];
    const float* Wa_w  = (const float*)d_in[4];
    const float* Wa_b  = (const float*)d_in[5];
    const float* Ua_w  = (const float*)d_in[6];
    const float* Ua_b  = (const float*)d_in[7];
    const float* Va_w  = (const float*)d_in[8];
    const float* Va_b  = (const float*)d_in[9];
    const float* W_ih  = (const float*)d_in[10];
    const float* W_hh  = (const float*)d_in[11];
    const float* b_ih  = (const float*)d_in[12];
    const float* b_hh  = (const float*)d_in[13];
    const float* out_w = (const float*)d_in[14];
    const float* out_b = (const float*)d_in[15];

    float* out  = (float*)d_out;
    float* dec  = out;                                   // [32,64,32000]
    float* hT   = out + (size_t)Bz * Sz * Vz;            // [1,32,512]
    float* attn = hT + (size_t)Bz * Hz;                  // [32,64,128]

    char* w = (char*)d_ws;
    float*  ua_keys = (float*)(w + 0);                              //  8 MB
    float*  hf0     = (float*)(w + 8388608);
    float*  hf1     = (float*)(w + 8454144);
    __bf16* hbf0    = (__bf16*)(w + 8519680);
    __bf16* hbf1    = (__bf16*)(w + 8552448);
    __bf16* x_bf    = (__bf16*)(w + 8585216);
    __bf16* h_seq   = (__bf16*)(w + 8650752);                       //  2 MB  [64,32,512]
    __bf16* Wih_bf  = (__bf16*)(w + 10747904);                      //  3 MB
    __bf16* Whh_bf  = (__bf16*)(w + 13893632);                      //  1.5 MB
    __bf16* outw_bf = (__bf16*)(w + 15466496);                      // 32.8 MB
    float*  hf[2]   = { hf0, hf1 };
    __bf16* hbf[2]  = { hbf0, hbf1 };

    // weight conversions + init + ua_keys
    k_cvt_bf16<<<2048, 256, 0, stream>>>(out_w, outw_bf, Vz * Hz);
    k_cvt_bf16<<<512, 256, 0, stream>>>(W_ih, Wih_bf, 3 * Hz * 2 * Hz);
    k_cvt_bf16<<<256, 256, 0, stream>>>(W_hh, Whh_bf, 3 * Hz * Hz);
    k_init_h<<<64, 256, 0, stream>>>(eh, hf[0], hbf[0]);
    k_uakeys<<<Bz * Tz, 256, 0, stream>>>(enc, Ua_w, Ua_b, ua_keys);

    // sequential recurrence
    for (int s = 0; s < Sz; ++s) {
        int cur = s & 1, nxt = cur ^ 1;
        k_attn<<<Bz, 256, 0, stream>>>(enc, ua_keys, hf[cur], Wa_w, Wa_b, Va_w, Va_b,
                                       emb, tgt, x_bf, attn, s);
        k_gru<<<Hz / 32, 128, 0, stream>>>(x_bf, hbf[cur], hf[cur], Wih_bf, Whh_bf,
                                           b_ih, b_hh, hf[nxt], hbf[nxt], h_seq, s);
    }

    // batched logits + log_softmax + final hidden
    k_logits<<<dim3(Vz / 64, (Sz * Bz) / 32), 256, 0, stream>>>(h_seq, outw_bf, out_b, dec);
    k_logsoftmax<<<Sz * Bz, 256, 0, stream>>>(dec);
    k_copy_f32<<<64, 256, 0, stream>>>(hf[0], hT, Bz * Hz);
}
